// GraphAttention_75857712382202
// MI455X (gfx1250) — matrix operand order repaired and checked by reading, not compiled, not run on hardware
//
#include <hip/hip_runtime.h>

// ---------------------------------------------------------------------------
// GAT forward, fused for MI455X (gfx1250, wave32, WMMA f16 16x16x32).
//
// Roofline: adj (268 MB) read exactly once from HBM (~11.5 us @ 23.3 TB/s);
// Whf16 (4 MB) + f2 (128 KB) are L2/WGP$-resident; 34 GFLOP of f16 WMMA is
// negligible vs matrix-core peak. Softmax is done in a single pass using the
// separable bound m_i = lrelu(f1_i + max_j f2_j) (LeakyReLU is monotone), so
// no second adj scan and no online-softmax accumulator rescaling.
// ---------------------------------------------------------------------------

typedef __attribute__((ext_vector_type(16))) _Float16 v16h;
typedef __attribute__((ext_vector_type(8)))  _Float16 v8h;
typedef __attribute__((ext_vector_type(8)))  float    v8f;

#define N_NODES 8192
#define C_IN    512
#define C_OUT   64
#define HEADS   4
#define ALPHA   0.2f
#define LOG2E   1.4426950408889634f

static __device__ __forceinline__ float lrelu(float x) {
    // x>=0 -> x ; x<0 -> 0.2x  (valid because 0 < alpha < 1)
    return fmaxf(x, ALPHA * x);
}

// ---------------------------------------------------------------------------
// Kernel 1: W [H][C_IN][C_OUT] f32 row-major  ->  Wf16 column-major
//           wcm[(h*C_OUT + c)*C_IN + k] = (f16) W[h][k][c]
// ---------------------------------------------------------------------------
__global__ void gat_prep_w(const float* __restrict__ W, _Float16* __restrict__ wcm) {
    int o = blockIdx.x * blockDim.x + threadIdx.x;
    if (o >= HEADS * C_OUT * C_IN) return;
    int h = o / (C_OUT * C_IN);
    int r = o % (C_OUT * C_IN);
    int c = r / C_IN;
    int k = r % C_IN;
    wcm[o] = (_Float16)W[(size_t)(h * C_IN + k) * C_OUT + c];
}

// ---------------------------------------------------------------------------
// Kernel 2: Wh = x @ W per head via V_WMMA_F32_16X16X32_F16.
// Block = 128 threads = 4 waves; wave w handles head w, 16-row tile i0.
// Also emits f1 = Wh.a1, f2 = Wh.a2 (f32) and Whf16 column-major [h][col][row].
// ---------------------------------------------------------------------------
__global__ __launch_bounds__(128) void gat_wh(
    const float*    __restrict__ x,
    const _Float16* __restrict__ wcm,
    const float*    __restrict__ a1,
    const float*    __restrict__ a2,
    _Float16*       __restrict__ whcm,
    float*          __restrict__ f1,
    float*          __restrict__ f2)
{
    const int lane = threadIdx.x & 31;
    const int h    = threadIdx.x >> 5;   // head = wave id
    const int i0   = blockIdx.x * 16;
    const int li   = lane & 15;
    const int hs   = lane >> 4;          // half-wave select

    float a1v[4], a2v[4];
#pragma unroll
    for (int nt = 0; nt < 4; ++nt) {
        a1v[nt] = a1[h * C_OUT + nt * 16 + li];
        a2v[nt] = a2[h * C_OUT + nt * 16 + li];
    }

    v8f acc[4];
#pragma unroll
    for (int nt = 0; nt < 4; ++nt) acc[nt] = (v8f)0.0f;

    const float* xrow = x + (size_t)(i0 + li) * C_IN;

    for (int kb = 0; kb < C_IN; kb += 32) {
        // A 16x32 f16: lane li = row M; K = kb + hs*8 + {0..7} and +16.
        const float* xp = xrow + kb + hs * 8;
        float4 g0a = *(const float4*)(xp);
        float4 g0b = *(const float4*)(xp + 4);
        float4 g1a = *(const float4*)(xp + 16);
        float4 g1b = *(const float4*)(xp + 20);
        v16h a;
        a[0]  = (_Float16)g0a.x; a[1]  = (_Float16)g0a.y;
        a[2]  = (_Float16)g0a.z; a[3]  = (_Float16)g0a.w;
        a[4]  = (_Float16)g0b.x; a[5]  = (_Float16)g0b.y;
        a[6]  = (_Float16)g0b.z; a[7]  = (_Float16)g0b.w;
        a[8]  = (_Float16)g1a.x; a[9]  = (_Float16)g1a.y;
        a[10] = (_Float16)g1a.z; a[11] = (_Float16)g1a.w;
        a[12] = (_Float16)g1b.x; a[13] = (_Float16)g1b.y;
        a[14] = (_Float16)g1b.z; a[15] = (_Float16)g1b.w;

#pragma unroll
        for (int nt = 0; nt < 4; ++nt) {
            // B 32x16 f16: lane li = N col; K = kb + hs*8 + {0..7} and +16 (col-major)
            const _Float16* bp =
                wcm + (size_t)(h * C_OUT + nt * 16 + li) * C_IN + kb + hs * 8;
            v16h b = __builtin_shufflevector(*(const v8h*)bp, *(const v8h*)(bp + 16), 0, 1, 2, 3, 4, 5, 6, 7, 8, 9, 10, 11, 12, 13, 14, 15);
            acc[nt] = __builtin_amdgcn_wmma_f32_16x16x32_f16(
                false, a, false, b, (short)0, acc[nt], false, false);
        }
    }

    // C layout: lane, vgpr v -> row M = i0 + hs*8 + v, col = nt*16 + li.
    // Store Whf16 column-major: rows consecutive in v -> one 16B store per nt.
#pragma unroll
    for (int nt = 0; nt < 4; ++nt) {
        v8h w;
#pragma unroll
        for (int v = 0; v < 8; ++v) w[v] = (_Float16)acc[nt][v];
        *(v8h*)(whcm + (size_t)(h * C_OUT + nt * 16 + li) * N_NODES + i0 + hs * 8) = w;
    }

    // f1/f2: row dot over 64 cols = sum over nt (in-lane) + 16-lane half reduce.
#pragma unroll
    for (int v = 0; v < 8; ++v) {
        float s1 = 0.0f, s2 = 0.0f;
#pragma unroll
        for (int nt = 0; nt < 4; ++nt) {
            s1 += acc[nt][v] * a1v[nt];
            s2 += acc[nt][v] * a2v[nt];
        }
#pragma unroll
        for (int off = 1; off < 16; off <<= 1) {
            s1 += __shfl_xor(s1, off, 32);
            s2 += __shfl_xor(s2, off, 32);
        }
        if (li == 0) {
            f1[h * N_NODES + i0 + hs * 8 + v] = s1;
            f2[h * N_NODES + i0 + hs * 8 + v] = s2;
        }
    }
}

// ---------------------------------------------------------------------------
// Kernel 3: F2max[h] = max_j f2[h][j]  (softmax stabilizer bound)
// ---------------------------------------------------------------------------
__global__ void gat_f2max(const float* __restrict__ f2, float* __restrict__ f2m) {
    __shared__ float red[256];
    const int h = blockIdx.x;
    float m = -3.0e38f;
    for (int i = threadIdx.x; i < N_NODES; i += 256)
        m = fmaxf(m, f2[h * N_NODES + i]);
    red[threadIdx.x] = m;
    __syncthreads();
    for (int s = 128; s > 0; s >>= 1) {
        if (threadIdx.x < s)
            red[threadIdx.x] = fmaxf(red[threadIdx.x], red[threadIdx.x + s]);
        __syncthreads();
    }
    if (threadIdx.x == 0) f2m[h] = red[0];
}

// ---------------------------------------------------------------------------
// Kernel 4: fused masked softmax + (attention @ Wh), single pass over adj.
// Block = 4 waves = 4 heads on the SAME 16-row tile (adj shared via WGP$).
// Per 32-column chunk: build p-tile (A, f16) in-register, 4 WMMAs into 64 cols.
// ---------------------------------------------------------------------------
__global__ __launch_bounds__(128) void gat_attn(
    const int*      __restrict__ adj,
    const _Float16* __restrict__ whcm,
    const float*    __restrict__ f1,
    const float*    __restrict__ f2,
    const float*    __restrict__ f2m,
    float*          __restrict__ out)
{
    const int lane = threadIdx.x & 31;
    const int h    = threadIdx.x >> 5;
    const int i0   = blockIdx.x * 16;
    const int li   = lane & 15;
    const int hs   = lane >> 4;
    const int i    = i0 + li;             // this lane's A-matrix row

    const float c1 = f1[h * N_NODES + i];
    const float m  = lrelu(c1 + f2m[h]); // >= true masked row max -> p in (0,1]

    const float*    f2h  = f2 + h * N_NODES;
    const int*      adjr = adj + (size_t)i * N_NODES;
    const _Float16* whh  = whcm + (size_t)h * C_OUT * N_NODES;

    v8f acc[4];
#pragma unroll
    for (int nt = 0; nt < 4; ++nt) acc[nt] = (v8f)0.0f;
    float lsum = 0.0f;

    for (int jb = 0; jb < N_NODES; jb += 32) {
        const int j0 = jb + hs * 8;       // K pattern: j0+{0..7}, j0+16+{0..7}
        float4 fa = *(const float4*)(f2h + j0);
        float4 fb = *(const float4*)(f2h + j0 + 4);
        float4 fc = *(const float4*)(f2h + j0 + 16);
        float4 fd = *(const float4*)(f2h + j0 + 20);
        int4   ba = *(const int4*)(adjr + j0);
        int4   bb = *(const int4*)(adjr + j0 + 4);
        int4   bc = *(const int4*)(adjr + j0 + 16);
        int4   bd = *(const int4*)(adjr + j0 + 20);

        const float fv[16] = {fa.x, fa.y, fa.z, fa.w, fb.x, fb.y, fb.z, fb.w,
                              fc.x, fc.y, fc.z, fc.w, fd.x, fd.y, fd.z, fd.w};
        const int   av[16] = {ba.x, ba.y, ba.z, ba.w, bb.x, bb.y, bb.z, bb.w,
                              bc.x, bc.y, bc.z, bc.w, bd.x, bd.y, bd.z, bd.w};

        v16h pa;
#pragma unroll
        for (int t = 0; t < 16; ++t) {
            float e = c1 + fv[t];
            float s = fmaxf(e, ALPHA * e);                     // LeakyReLU
            float p = (av[t] > 0) ? exp2f((s - m) * LOG2E) : 0.0f;
            lsum += p;
            pa[t] = (_Float16)p;
        }

#pragma unroll
        for (int nt = 0; nt < 4; ++nt) {
            // B 32x16 f16 from column-major Whf16: K = jb + hs*8 + {0..7} and +16.
            const _Float16* bp =
                whh + (size_t)(nt * 16 + li) * N_NODES + jb + hs * 8;
            v16h b = __builtin_shufflevector(*(const v8h*)bp, *(const v8h*)(bp + 16), 0, 1, 2, 3, 4, 5, 6, 7, 8, 9, 10, 11, 12, 13, 14, 15);
            acc[nt] = __builtin_amdgcn_wmma_f32_16x16x32_f16(
                false, pa, false, b, (short)0, acc[nt], false, false);
        }
    }

    // Row sum: row i handled by lanes i and i+16 -> pairwise xor-16 reduce.
    const float ltot = lsum + __shfl_xor(lsum, 16, 32);

#pragma unroll
    for (int v = 0; v < 8; ++v) {
        // C row M = hs*8 + v ; its sum lives in lane (hs*8 + v).
        float lv = __shfl(ltot, hs * 8 + v, 32);
        float rv = 1.0f / lv;
        const int row = i0 + hs * 8 + v;
#pragma unroll
        for (int nt = 0; nt < 4; ++nt) {
            out[(size_t)row * (HEADS * C_OUT) + h * C_OUT + nt * 16 + li] =
                acc[nt][v] * rv;
        }
    }
}

// ---------------------------------------------------------------------------
// Launcher. Inputs: x[8192*512] f32, adj[8192*8192] i32, W[4*512*64] f32,
//                   a1[4*64] f32, a2[4*64] f32.  Output: [8192, 256] f32.
// Workspace layout (bytes):
//   [0,       256K)  Wf16 col-major
//   [256K,   4.25M)  Whf16 col-major
//   [4.25M, 4.375M)  f1
//   [4.375M, 4.5M)   f2
//   [4.5M,  +16B)    F2max
// ---------------------------------------------------------------------------
extern "C" void kernel_launch(void* const* d_in, const int* in_sizes, int n_in,
                              void* d_out, int out_size, void* d_ws, size_t ws_size,
                              hipStream_t stream) {
    const float* x   = (const float*)d_in[0];
    const int*   adj = (const int*)d_in[1];
    const float* W   = (const float*)d_in[2];
    const float* a1  = (const float*)d_in[3];
    const float* a2  = (const float*)d_in[4];
    float*       out = (float*)d_out;

    char* ws = (char*)d_ws;
    _Float16* wcm  = (_Float16*)(ws);
    _Float16* whcm = (_Float16*)(ws + (size_t)262144);
    float*    f1   = (float*)(ws + (size_t)262144 + 4194304);
    float*    f2   = (float*)(ws + (size_t)262144 + 4194304 + 131072);
    float*    f2m  = (float*)(ws + (size_t)262144 + 4194304 + 262144);

    gat_prep_w<<<(HEADS * C_OUT * C_IN + 255) / 256, 256, 0, stream>>>(W, wcm);
    gat_wh<<<N_NODES / 16, 128, 0, stream>>>(x, wcm, a1, a2, whcm, f1, f2);
    gat_f2max<<<HEADS, 256, 0, stream>>>(f2, f2m);
    gat_attn<<<N_NODES / 16, 128, 0, stream>>>(adj, whcm, f1, f2, f2m, out);
}
